// StoichKineticsODE_70798240907373
// MI455X (gfx1250) — matrix operand order repaired and checked by reading, not compiled, not run on hardware
//
#include <hip/hip_runtime.h>
#include <math.h>

#define N_MET   100000
#define N_RXN   50000
#define MSG_DIM 16
#define HIDDEN  32

typedef __attribute__((ext_vector_type(16))) _Float16 v16h;
typedef __attribute__((ext_vector_type(8)))  float    v8f;

// ---------------------------------------------------------------------------
// Kernel 0: zero the accumulation buffers (h_rxn in ws, dxdt region of out)
// ---------------------------------------------------------------------------
__global__ void sk_init_kernel(float* __restrict__ h_rxn, float* __restrict__ dxdt)
{
    int i = blockIdx.x * blockDim.x + threadIdx.x;
    if (i < N_RXN * MSG_DIM) h_rxn[i] = 0.0f;
    if (i < N_MET)           dxdt[i]  = 0.0f;
}

// ---------------------------------------------------------------------------
// Kernel 1: edge MLP (2 -> tanh 32 -> 16); the 32x16 layer runs on
// v_wmma_f32_16x16x32_f16 (A = 16 edges x 32 hidden, B = W2m).
// FULL TILES ONLY (n_full = e_sub/16): no per-lane guards, no ragged path,
// EXEC is all-1s throughout. Grid-stride loop; layer-1 weights and the B
// fragment live in registers for the whole wave lifetime.
// ---------------------------------------------------------------------------
__global__ void __launch_bounds__(256)
sk_edge_kernel(const float* __restrict__ x_met,
               const float* __restrict__ sto_sub,
               const int*   __restrict__ met_sub,
               const int*   __restrict__ rxn_sub,
               const float* __restrict__ W1m,   // [2][32]
               const float* __restrict__ b1m,   // [32]
               const float* __restrict__ W2m,   // [32][16]
               const float* __restrict__ b2m,   // [16]
               float*       __restrict__ h_rxn, // [N_RXN][16]
               int n_full)                      // number of full 16-edge tiles
{
    const int lane   = threadIdx.x & 31;
    const int half   = lane >> 4;            // which K-half this lane supplies
    const int m      = lane & 15;            // edge-in-tile (A row) == B column
    const int wave   = (blockIdx.x * blockDim.x + threadIdx.x) >> 5;
    const int nwaves = (gridDim.x * blockDim.x) >> 5;
    const int elast  = n_full * 16 - 1;      // clamp target for prefetch

    // ---- layer-1 weights for this lane's 16 K slots (registers, loaded once)
    // A 16x32 f16 layout: lanes 0-15 (half=0): VGPR j<4 -> K=2j..2j+1,
    // VGPR j>=4 -> K=16+2(j-4)..; lanes 16-31 (half=1): same +8.
    float w1x[16], w1s[16], bb1[16];
    #pragma unroll
    for (int j = 0; j < 8; ++j) {
        const int kbase = ((j >> 2) << 4) + (half << 3) + ((j & 3) << 1);
        #pragma unroll
        for (int t = 0; t < 2; ++t) {
            const int k = kbase + t;
            w1x[2 * j + t] = W1m[k];
            w1s[2 * j + t] = W1m[HIDDEN + k];
            bb1[2 * j + t] = b1m[k];
        }
    }
    // ---- B fragment (W2m column m) once per wave.
    // B 32x16 f16 layout: same K slots as A (VGPR j<4 -> K=half*8+2j.., j>=4 -> K=16+half*8+2(j-4)..) for N=m.
    v16h B;
    #pragma unroll
    for (int j = 0; j < 8; ++j) {
        const int k = ((j >> 2) << 4) + (half << 3) + ((j & 3) << 1);
        B[2 * j]     = (_Float16)W2m[k       * MSG_DIM + m];
        B[2 * j + 1] = (_Float16)W2m[(k + 1) * MSG_DIM + m];
    }
    const float bn = b2m[m];

    for (int tile = wave; tile < n_full; tile += nwaves) {
        const int edge0 = tile << 4;
        const int e     = edge0 + m;

        // branchless prefetch of the next tile's edge stream
        int ne = ((tile + nwaves) << 4) + m;
        ne = (ne > elast) ? elast : ne;
        __builtin_prefetch(&sto_sub[ne], 0, 0);
        __builtin_prefetch(&met_sub[ne], 0, 0);
        __builtin_prefetch(&rxn_sub[ne], 0, 0);

        const float x = x_met[met_sub[e]];
        const float s = sto_sub[e];

        // layer 1: 16 hidden activations at this lane's A slots
        v16h A;
        #pragma unroll
        for (int i = 0; i < 16; ++i)
            A[i] = (_Float16)tanhf(fmaf(x, w1x[i], fmaf(s, w1s[i], bb1[i])));

        v8f C = {};
        C = __builtin_amdgcn_wmma_f32_16x16x32_f16(
                /*neg_a=*/false, A, /*neg_b=*/false, B,
                /*c_mod=*/(short)0, C, /*reuse_a=*/false, /*reuse_b=*/false);

        // C layout: VGPR v -> edge row v + half*8, msg dim N = m.
        const int  rowbase = edge0 + (half << 3);
        const int4 r0 = *(const int4*)(rxn_sub + rowbase);
        const int4 r1 = *(const int4*)(rxn_sub + rowbase + 4);
        const int  ri[8] = {r0.x, r0.y, r0.z, r0.w, r1.x, r1.y, r1.z, r1.w};
        #pragma unroll
        for (int v = 0; v < 8; ++v)
            atomicAdd(&h_rxn[ri[v] * MSG_DIM + m], C[v] + bn);
    }
}

// ---------------------------------------------------------------------------
// Kernel 1b: ragged tail (< 16 edges), plain per-edge VALU path.
// Launched only when e_sub % 16 != 0 (never for E_SUB = 2M).
// ---------------------------------------------------------------------------
__global__ void sk_edge_tail_kernel(const float* __restrict__ x_met,
                                    const float* __restrict__ sto_sub,
                                    const int*   __restrict__ met_sub,
                                    const int*   __restrict__ rxn_sub,
                                    const float* __restrict__ W1m,
                                    const float* __restrict__ b1m,
                                    const float* __restrict__ W2m,
                                    const float* __restrict__ b2m,
                                    float*       __restrict__ h_rxn,
                                    int e_start, int e_sub)
{
    const int e = e_start + blockIdx.x * blockDim.x + threadIdx.x;
    if (e >= e_sub) return;
    const float x = x_met[met_sub[e]];
    const float s = sto_sub[e];
    float hid[HIDDEN];
    #pragma unroll
    for (int k = 0; k < HIDDEN; ++k)
        hid[k] = tanhf(fmaf(x, W1m[k], fmaf(s, W1m[HIDDEN + k], b1m[k])));
    const int r = rxn_sub[e];
    #pragma unroll
    for (int n = 0; n < MSG_DIM; ++n) {
        float acc = b2m[n];
        #pragma unroll
        for (int k = 0; k < HIDDEN; ++k)
            acc = fmaf(hid[k], W2m[k * MSG_DIM + n], acc);
        atomicAdd(&h_rxn[r * MSG_DIM + n], acc);
    }
}

// ---------------------------------------------------------------------------
// Kernel 2: per-reaction rate MLP (16 -> tanh 32 -> 1) + softplus.
// Only ~54 MFLOP total -> scalar VALU, one thread per reaction.
// ---------------------------------------------------------------------------
__global__ void __launch_bounds__(256)
sk_rate_kernel(const float* __restrict__ h_rxn, // [N_RXN][16]
               const float* __restrict__ W1r,   // [16][32]
               const float* __restrict__ b1r,   // [32]
               const float* __restrict__ W2r,   // [32]
               const float* __restrict__ b2r,   // [1]
               float*       __restrict__ v_out) // [N_RXN]
{
    const int r = blockIdx.x * blockDim.x + threadIdx.x;
    if (r >= N_RXN) return;

    const float4* hp = (const float4*)(h_rxn + r * MSG_DIM);
    const float4 h0 = hp[0], h1 = hp[1], h2 = hp[2], h3 = hp[3];
    const float h[MSG_DIM] = {h0.x, h0.y, h0.z, h0.w, h1.x, h1.y, h1.z, h1.w,
                              h2.x, h2.y, h2.z, h2.w, h3.x, h3.y, h3.z, h3.w};

    float acc = b2r[0];
    #pragma unroll 4
    for (int j = 0; j < HIDDEN; ++j) {
        float z = b1r[j];
        #pragma unroll
        for (int i = 0; i < MSG_DIM; ++i) z = fmaf(h[i], W1r[i * HIDDEN + j], z);
        acc = fmaf(tanhf(z), W2r[j], acc);
    }
    // numerically stable softplus: max(x,0) + log1p(exp(-|x|))
    v_out[r] = fmaxf(acc, 0.0f) + log1pf(expf(-fabsf(acc)));
}

// ---------------------------------------------------------------------------
// Kernel 3: dxdt scatter, 4 edges per thread (vector loads of the 48 MB
// edge stream): dxdt[met_all[e]] += sto_all[e] * v[rxn_all[e]]
// ---------------------------------------------------------------------------
__global__ void __launch_bounds__(256)
sk_scatter_kernel(const float* __restrict__ sto_all,
                  const int*   __restrict__ met_all,
                  const int*   __restrict__ rxn_all,
                  const float* __restrict__ v_in,
                  float*       __restrict__ dxdt,
                  int e_all)
{
    const int e0 = (blockIdx.x * blockDim.x + threadIdx.x) << 2;
    if (e0 + 4 <= e_all) {
        const float4 s  = *(const float4*)(sto_all + e0);
        const int4   mi = *(const int4*)(met_all + e0);
        const int4   ri = *(const int4*)(rxn_all + e0);
        atomicAdd(&dxdt[mi.x], s.x * v_in[ri.x]);
        atomicAdd(&dxdt[mi.y], s.y * v_in[ri.y]);
        atomicAdd(&dxdt[mi.z], s.z * v_in[ri.z]);
        atomicAdd(&dxdt[mi.w], s.w * v_in[ri.w]);
    } else {
        for (int e = e0; e < e_all; ++e)
            atomicAdd(&dxdt[met_all[e]], sto_all[e] * v_in[rxn_all[e]]);
    }
}

// ---------------------------------------------------------------------------
extern "C" void kernel_launch(void* const* d_in, const int* in_sizes, int n_in,
                              void* d_out, int out_size, void* d_ws, size_t ws_size,
                              hipStream_t stream)
{
    // setup_inputs() dict order:
    const float* x_met   = (const float*)d_in[0];   // [N_MET]
    const float* sto_sub = (const float*)d_in[1];   // [E_SUB]
    const float* sto_all = (const float*)d_in[2];   // [E_ALL]
    const float* W1m     = (const float*)d_in[3];   // [2,32]
    const float* b1m     = (const float*)d_in[4];   // [32]
    const float* W2m     = (const float*)d_in[5];   // [32,16]
    const float* b2m     = (const float*)d_in[6];   // [16]
    const float* W1r     = (const float*)d_in[7];   // [16,32]
    const float* b1r     = (const float*)d_in[8];   // [32]
    const float* W2r     = (const float*)d_in[9];   // [32,1]
    const float* b2r     = (const float*)d_in[10];  // [1]
    const int*   met_sub = (const int*)d_in[11];    // [E_SUB]
    const int*   rxn_sub = (const int*)d_in[12];    // [E_SUB]
    const int*   met_all = (const int*)d_in[13];    // [E_ALL]
    const int*   rxn_all = (const int*)d_in[14];    // [E_ALL]

    const int e_sub = in_sizes[1];
    const int e_all = in_sizes[2];

    float* dxdt  = (float*)d_out;           // [N_MET]
    float* v_out = (float*)d_out + N_MET;   // [N_RXN]
    float* h_rxn = (float*)d_ws;            // [N_RXN * MSG_DIM] = 3.2 MB

    // 0) zero accumulators
    {
        const int n = N_RXN * MSG_DIM;      // > N_MET
        sk_init_kernel<<<(n + 255) / 256, 256, 0, stream>>>(h_rxn, dxdt);
    }
    // 1) edge MLP + WMMA + scatter into h_rxn (full 16-edge tiles only)
    {
        const int n_full = e_sub >> 4;                      // 125000
        if (n_full > 0) {
            int blocks = (n_full + 7) / 8;                  // 8 waves/block
            if (blocks > 2048) blocks = 2048;               // ~8 tiles per wave
            sk_edge_kernel<<<blocks, 256, 0, stream>>>(
                x_met, sto_sub, met_sub, rxn_sub, W1m, b1m, W2m, b2m,
                h_rxn, n_full);
        }
        const int tail = e_sub - (n_full << 4);             // 0 for E_SUB = 2M
        if (tail > 0) {
            sk_edge_tail_kernel<<<1, 32, 0, stream>>>(
                x_met, sto_sub, met_sub, rxn_sub, W1m, b1m, W2m, b2m,
                h_rxn, n_full << 4, e_sub);
        }
    }
    // 2) rate MLP + softplus -> v
    sk_rate_kernel<<<(N_RXN + 255) / 256, 256, 0, stream>>>(
        h_rxn, W1r, b1r, W2r, b2r, v_out);
    // 3) dxdt scatter (4 edges/thread)
    {
        const int threads = (e_all + 3) / 4;
        sk_scatter_kernel<<<(threads + 255) / 256, 256, 0, stream>>>(
            sto_all, met_all, rxn_all, v_out, dxdt, e_all);
    }
}
